// InfiniAttention_44418551775469
// MI455X (gfx1250) — matrix operand order repaired and checked by reading, not compiled, not run on hardware
//
#include <hip/hip_runtime.h>
#include <hip/hip_bf16.h>

typedef float  v2f   __attribute__((ext_vector_type(2)));
typedef float  v8f   __attribute__((ext_vector_type(8)));
typedef __bf16 v16bf __attribute__((ext_vector_type(16)));

#define SB 4
#define HH 32
#define SS 2048
#define DD 128
#define DM 4096   // H*D
#define BH (SB*HH)

// elu(x)+1 : x>=0 -> x+1 ; x<0 -> exp(x)
__device__ __forceinline__ float phi1(float x) { return x > 0.f ? x + 1.f : __expf(x); }
__device__ __forceinline__ float sigm(float x) { return 1.f / (1.f + __expf(-x)); }
__device__ __forceinline__ unsigned short bf16rne(float f) {
    unsigned int u = __float_as_uint(f);
    u += 0x7FFFu + ((u >> 16) & 1u);     // round-to-nearest-even
    return (unsigned short)(u >> 16);
}

// ---------------------------------------------------------------------------
// Kernel 1: split-K partials of  memory = phi(K)^T @ V  (f32 WMMA, 16x16x4).
// 4 segments of S per (b,h) -> 512 blocks. Normalization is folded into the
// A-fragment build (each phi(K) value appears exactly once per wave) and
// reduced across lane-halves with a wave32 shfl_xor — no second pass over K.
// ---------------------------------------------------------------------------
__global__ void k_memory(const float* __restrict__ K, const float* __restrict__ V,
                         float* __restrict__ memP, float* __restrict__ normP) {
    const int bx   = blockIdx.x;
    const int bh   = bx >> 2, seg = bx & 3;   // bh = b*32+h, seg = S quarter
    const int tid  = threadIdx.x;
    const int wave = tid >> 5, lane = tid & 31;
    const int half = lane >> 4, l = lane & 15;
    const float* Kb = K + (size_t)bh * SS * DD;
    const float* Vb = V + (size_t)bh * SS * DD;
    const int m0   = wave * 16;               // this wave's key-dim tile
    const int sBeg = seg * (SS / 4), sEnd = sBeg + (SS / 4);

    v8f acc[8] = {};
    float nacc = 0.f;                         // partial sum_s phi(K[s, m0+l])
    for (int s0 = sBeg; s0 < sEnd; s0 += 4) {
        const int ra = s0 + 2 * half;         // A: lanes0-15 K=0,1 ; lanes16-31 K=2,3
        v2f a;
        a.x = phi1(Kb[(size_t)ra       * DD + m0 + l]);
        a.y = phi1(Kb[(size_t)(ra + 1) * DD + m0 + l]);
        nacc += a.x + a.y;
#pragma unroll
        for (int nt = 0; nt < 8; ++nt) {
            v2f bv;                           // B 4x16: same half/K striping, N=lane&15
            bv.x = Vb[(size_t)ra       * DD + nt * 16 + l];
            bv.y = Vb[(size_t)(ra + 1) * DD + nt * 16 + l];
            acc[nt] = __builtin_amdgcn_wmma_f32_16x16x4_f32(
                false, a, false, bv, (short)0, acc[nt], false, false);
        }
    }
    float* Mo = memP + ((size_t)seg * BH + bh) * (DD * DD);
#pragma unroll
    for (int nt = 0; nt < 8; ++nt)
#pragma unroll
        for (int r = 0; r < 8; ++r)           // D: VGPR r -> M=r (lo half) / 8+r (hi half)
            Mo[(size_t)(m0 + r + 8 * half) * DD + nt * 16 + l] = acc[nt][r];

    nacc += __shfl_xor(nacc, 16);             // combine halves (s%4 in {0,1} / {2,3}) (wave32)
    if (half == 0)
        normP[((size_t)seg * BH + bh) * DD + m0 + l] = nacc;
}

// ---------------------------------------------------------------------------
// Kernel 1b: reduce the 4 split-K partials (float4-vectorized).
// ---------------------------------------------------------------------------
__global__ void k_reduce(const float* __restrict__ memP, const float* __restrict__ normP,
                         float* __restrict__ memW, float* __restrict__ normW) {
    const int i = blockIdx.x * 256 + threadIdx.x;
    const int MEMQ = BH * DD * DD / 4;        // 524288 float4s of memory
    const int NQ   = BH * DD / 4;             // 4096 float4s of norm
    if (i < MEMQ) {
        const float4* p = (const float4*)memP;
        float4 a = p[i], b = p[i + MEMQ], c = p[i + 2 * MEMQ], d = p[i + 3 * MEMQ];
        float4 r; r.x = a.x + b.x + c.x + d.x; r.y = a.y + b.y + c.y + d.y;
        r.z = a.z + b.z + c.z + d.z; r.w = a.w + b.w + c.w + d.w;
        ((float4*)memW)[i] = r;
    } else {
        const int j = i - MEMQ;               // < 4096
        const float4* p = (const float4*)normP;
        float4 a = p[j], b = p[j + NQ], c = p[j + 2 * NQ], d = p[j + 3 * NQ];
        float4 r; r.x = a.x + b.x + c.x + d.x; r.y = a.y + b.y + c.y + d.y;
        r.z = a.z + b.z + c.z + d.z; r.w = a.w + b.w + c.w + d.w;
        ((float4*)normW)[j] = r;
    }
}

// ---------------------------------------------------------------------------
// Kernel 2: per (b,h,s-tile of 128): retrieved = phi(Q) @ memory (f32 WMMA),
// denom = phi(Q)·norm, gate, add local, emit bf16 attn in [B,S,H*D] layout.
// phi(Q) tile staged in LDS (b128 global loads) with stride 129 ->
// conflict-free A-fragment reads.
// ---------------------------------------------------------------------------
__global__ void k_retrieve(const float* __restrict__ Q, const float* __restrict__ L,
                           const float* __restrict__ bal,
                           const float* __restrict__ memW, const float* __restrict__ normW,
                           unsigned short* __restrict__ attnW) {
    extern __shared__ float sm[];
    float* qs = sm;                 // 128*129
    float* ns = sm + 128 * 129;     // 128
    float* dn = ns + 128;           // 128
    const int bx = blockIdx.x;
    const int st = bx & 15, bh = bx >> 4;
    const int b = bh >> 5, h = bh & 31;
    const int sbase = st * 128;
    const int tid = threadIdx.x;

    const float* Qb = Q + ((size_t)bh * SS + sbase) * DD;
    for (int i = tid * 4; i < 128 * 128; i += 256 * 4) {
        const int r = i >> 7, c = i & 127;
        float4 q = *(const float4*)(Qb + (size_t)r * DD + c);
        float* qp = qs + r * 129 + c;
        qp[0] = phi1(q.x); qp[1] = phi1(q.y); qp[2] = phi1(q.z); qp[3] = phi1(q.w);
    }
    if (tid < 128) ns[tid] = normW[(size_t)bh * DD + tid];
    __syncthreads();
    if (tid < 128) {
        float s = 0.f;
        for (int k = 0; k < 128; ++k) s += qs[tid * 129 + k] * ns[k];
        dn[tid] = s;
    }
    __syncthreads();

    const int wave = tid >> 5, lane = tid & 31, half = lane >> 4, l = lane & 15;
    const int m0 = wave * 16;
    v8f acc[8] = {};
    const float* Mb = memW + (size_t)bh * DD * DD;
    for (int k0 = 0; k0 < DD; k0 += 4) {
        const int c = k0 + 2 * half;
        v2f a;
        a.x = qs[(m0 + l) * 129 + c];
        a.y = qs[(m0 + l) * 129 + c + 1];
#pragma unroll
        for (int nt = 0; nt < 8; ++nt) {
            v2f bv;
            bv.x = Mb[(size_t)c       * DD + nt * 16 + l];
            bv.y = Mb[(size_t)(c + 1) * DD + nt * 16 + l];
            acc[nt] = __builtin_amdgcn_wmma_f32_16x16x4_f32(
                false, a, false, bv, (short)0, acc[nt], false, false);
        }
    }
    const float gw = sigm(bal[h]);
    const float lw = sigm(1.f - bal[h]);
    const float* Lb = L + ((size_t)bh * SS + sbase) * DD;
#pragma unroll
    for (int nt = 0; nt < 8; ++nt)
#pragma unroll
        for (int r = 0; r < 8; ++r) {
            const int rl  = m0 + r + 8 * half;     // row inside 128-row tile
            const int col = nt * 16 + l;           // d inside head
            float v = gw * acc[nt][r] / dn[rl] + lw * Lb[(size_t)rl * DD + col];
            attnW[((size_t)b * SS + sbase + rl) * DM + h * DD + col] = bf16rne(v);
        }
}

// ---------------------------------------------------------------------------
// Kernel 3: convert w_o (f32 [4096,4096]) -> bf16, pre-swizzled into the exact
// 32x16 bf16 B-fragment layout: tile(KT,NT) = 32 lanes x 16 bf16 contiguous.
// lane n%16      holds w_o[n][K0..K0+15]   (K pairs packed per VGPR)
// lane 16+(n%16) holds w_o[n][K0+16..K0+31]
// ---------------------------------------------------------------------------
__global__ void k_pack_wo(const float* __restrict__ wo, unsigned short* __restrict__ wt) {
    const int gid = blockIdx.x * 256 + threadIdx.x;  // 4096*4096/16 groups
    const int n   = gid >> 8;                        // output-feature row of w_o
    const int k16 = (gid & 255) << 4;                // 16-element K chunk
    const float* src = wo + (size_t)n * DM + k16;
    unsigned short tmp[16];
#pragma unroll
    for (int i = 0; i < 16; ++i) tmp[i] = bf16rne(src[i]);
    const int KT   = k16 >> 5;
    const int lane = (n & 15) + (((k16 >> 4) & 1) << 4);
    const int NT   = n >> 4;
    unsigned short* dst = wt + ((size_t)(KT * (DM / 16) + NT) * 512) + lane * 16;
    ((uint4*)dst)[0] = ((uint4*)tmp)[0];
    ((uint4*)dst)[1] = ((uint4*)tmp)[1];
}

// ---------------------------------------------------------------------------
// Kernel 4: o_proj  out[8192,4096] = attn_bf16 @ w_o^T  via
// v_wmma_f32_16x16x32_bf16. A-fragments are contiguous 16B chunks of the
// row-major bf16 attn; B-fragments are contiguous 32B lane chunks of wt.
// ---------------------------------------------------------------------------
__global__ void k_oproj(const unsigned short* __restrict__ attnW,
                        const unsigned short* __restrict__ wt,
                        float* __restrict__ out) {
    const int bx  = blockIdx.x;
    const int ntb = bx & 31, mtb = bx >> 5;
    const int tid = threadIdx.x;
    const int wave = tid >> 5, lane = tid & 31, half = lane >> 4, l = lane & 15;
    const int M0 = mtb * 128 + wave * 16;
    const int N0 = ntb * 128;
    v8f acc[8] = {};
    const unsigned short* arow = attnW + (size_t)(M0 + l) * DM;
    for (int K0 = 0; K0 < DM; K0 += 32) {
        union { uint4 q[2]; v16bf v; } A;
        // A 16x32, this code's contiguous k order (as in wt): lanes0-15 K={0..15}, lanes16-31 K={16..31}
        A.q[0] = *(const uint4*)(arow + K0 + 16 * half);
        A.q[1] = *(const uint4*)(arow + K0 + 16 * half + 8);
        if (K0 + 32 < DM) __builtin_prefetch(arow + K0 + 32, 0, 0);
#pragma unroll
        for (int nt = 0; nt < 8; ++nt) {
            const unsigned short* bp =
                wt + ((size_t)((K0 >> 5) * (DM / 16) + (ntb * 8 + nt)) * 512) + lane * 16;
            union { uint4 q[2]; v16bf v; } B;
            B.q[0] = ((const uint4*)bp)[0];
            B.q[1] = ((const uint4*)bp)[1];
            acc[nt] = __builtin_amdgcn_wmma_f32_16x16x32_bf16(
                false, A.v, false, B.v, (short)0, acc[nt], false, false);
        }
    }
#pragma unroll
    for (int nt = 0; nt < 8; ++nt)
#pragma unroll
        for (int r = 0; r < 8; ++r)
            out[(size_t)(M0 + r + 8 * half) * DM + N0 + nt * 16 + l] = acc[nt][r];
}

extern "C" void kernel_launch(void* const* d_in, const int* in_sizes, int n_in,
                              void* d_out, int out_size, void* d_ws, size_t ws_size,
                              hipStream_t stream) {
    (void)in_sizes; (void)n_in; (void)out_size; (void)ws_size;
    const float* Q   = (const float*)d_in[0];
    const float* K   = (const float*)d_in[1];
    const float* V   = (const float*)d_in[2];
    const float* L   = (const float*)d_in[3];
    const float* bal = (const float*)d_in[4];
    const float* wo  = (const float*)d_in[5];

    char* ws = (char*)d_ws;
    float*          memW  = (float*)ws;                                   // 8 MB
    float*          normW = (float*)(ws + 8388608);                       // 64 KB
    unsigned short* attnW = (unsigned short*)(ws + 8454144);              // 64 MB bf16
    unsigned short* wtW   = (unsigned short*)(ws + 8454144 + 67108864);   // 32 MB bf16
    // split-K partials live inside the (not-yet-written) attn region: they are
    // dead before k_retrieve stores attn there.  32 MB + 256 KB < 64 MB.
    float* memP  = (float*)(ws + 8454144);
    float* normP = (float*)(ws + 8454144 + 33554432);
    float* out = (float*)d_out;

    k_pack_wo <<<4096, 256, 0, stream>>>(wo, wtW);
    k_memory  <<<BH * 4, 256, 0, stream>>>(K, V, memP, normP);
    k_reduce  <<<(BH * DD * DD / 4 + BH * DD / 4) / 256, 256, 0, stream>>>(memP, normP, memW, normW);
    const size_t smem = (128 * 129 + 256) * sizeof(float);                // 67,072 B
    k_retrieve<<<BH * (SS / 128), 256, smem, stream>>>(Q, L, bal, memW, normW, attnW);
    k_oproj   <<<(SB * SS / 128) * (DM / 128), 256, 0, stream>>>(attnW, wtW, out);
}
